// VectorQuantizer_10763188044254
// MI455X (gfx1250) — matrix operand order repaired and checked by reading, not compiled, not run on hardware
//
#include <hip/hip_runtime.h>
#include <hip/hip_bf16.h>
#include <math.h>

typedef __attribute__((ext_vector_type(16))) _Float16 v16h;
typedef __attribute__((ext_vector_type(8)))  float    v8f;

#define K_CODES 512
#define DIM 64
#define N_ROWS 65536            // 64 * 32 * 32
#define N_TILES 32              // 512 / 16
#define OUT_NCHW_OFF 1
#define OUT_PERP_OFF (1 + 4194304)
#define OUT_NHWC_OFF (2 + 4194304)
#define TOTAL_ELEMS 4194304.0f  // 64*64*32*32

// d_ws layout (bytes):
//   [0,      65536) : B fragments, f16: [32 tiles][2 khalf][32 lanes][16 halves]
//   [65536,  67584) : e_norm f32 [512]
//   [67584,  69632) : histogram u32 [512]
//   [69632,  69636) : loss accumulator f32

union AFrag { v16h v; _Float16 h[16]; };
union CFrag { v8f  v; float    f[8];  };

// ---------------- prep: ||e||^2, zero hist/loss, swizzle codebook to WMMA B layout
__global__ void vq_prep(const float* __restrict__ emb, void* ws) {
  _Float16*     bbuf    = (_Float16*)ws;
  float*        enorm   = (float*)((char*)ws + 65536);
  unsigned int* hist    = (unsigned int*)((char*)ws + 67584);
  float*        lossacc = (float*)((char*)ws + 69632);
  int tid = blockIdx.x * blockDim.x + threadIdx.x;
  if (tid < K_CODES) {
    const float* row = emb + tid * DIM;
    float s = 0.f;
    for (int d = 0; d < DIM; ++d) s += row[d] * row[d];
    enorm[tid] = s;
    hist[tid]  = 0u;
    if (tid == 0) *lossacc = 0.f;
  }
  int bt = tid - K_CODES;                 // [0, 2048): one (tile,khalf,lane) slot
  if (bt >= 0 && bt < 2048) {
    int lane = bt & 31;
    int kh   = (bt >> 5) & 1;
    int tile = bt >> 6;
    int code = tile * 16 + (lane & 15);   // B-matrix: N = lane % 16
    int dbase = kh * 32 + ((lane >= 16) ? 8 : 0);  // lanes 0-15: K {0..7,16..23}, 16-31: K {8..15,24..31}
    const float* row = emb + (size_t)code * DIM + dbase;
    _Float16* dst = bbuf + (size_t)bt * 16;
    for (int i = 0; i < 16; ++i) dst[i] = (_Float16)row[i + ((i >= 8) ? 8 : 0)];
  }
}

// ---------------- main: WMMA distance GEMM + argmin + gather + outputs + loss/hist
__global__ void __launch_bounds__(128) vq_main(const float* __restrict__ x,
                                               const float* __restrict__ emb,
                                               float* __restrict__ out, void* ws) {
  const _Float16*     bbuf    = (const _Float16*)ws;
  const float*        enorm   = (const float*)((const char*)ws + 65536);
  unsigned int*       hist    = (unsigned int*)((char*)ws + 67584);
  float*              lossacc = (float*)((char*)ws + 69632);

  const int lane = threadIdx.x & 31;
  const int wave = threadIdx.x >> 5;
  const int rt   = blockIdx.x * 4 + wave;   // row-tile index [0, 4096)
  const int row0 = rt * 16;

  // ---- A fragment: rows row0..row0+15, f32 NCHW -> f16 WMMA 16x32 layout.
  // lane<16 holds K in {0..7, 16..23}; lane>=16 holds K in {8..15, 24..31}.
  const int m       = lane & 15;
  const int n_lane  = row0 + m;
  const int b_lane  = n_lane >> 10;
  const int hw_lane = n_lane & 1023;
  const float* xb   = x + (size_t)b_lane * 65536 + hw_lane;  // + d*1024
  const int doff    = (lane >= 16) ? 8 : 0;
  AFrag a0, a1;
  #pragma unroll
  for (int i = 0; i < 16; ++i) {
    int d = doff + i + ((i >= 8) ? 8 : 0);
    a0.h[i] = (_Float16)xb[(size_t)d * 1024];
    a1.h[i] = (_Float16)xb[(size_t)(d + 32) * 1024];
  }

  float minval[8];
  int   minidx[8];
  #pragma unroll
  for (int j = 0; j < 8; ++j) { minval[j] = 3.4e38f; minidx[j] = 0; }

  for (int t = 0; t < N_TILES; ++t) {
    const v16h b0 = *(const v16h*)(bbuf + ((size_t)(t * 2 + 0) * 32 + lane) * 16);
    const v16h b1 = *(const v16h*)(bbuf + ((size_t)(t * 2 + 1) * 32 + lane) * 16);
    if (t + 1 < N_TILES)
      __builtin_prefetch(bbuf + ((size_t)((t + 1) * 2) * 32 + lane) * 16, 0, 0);
    CFrag c; c.v = (v8f){};
    c.v = __builtin_amdgcn_wmma_f32_16x16x32_f16(false, a0.v, false, b0,
                                                 (short)0, c.v, false, false);
    c.v = __builtin_amdgcn_wmma_f32_16x16x32_f16(false, a1.v, false, b1,
                                                 (short)0, c.v, false, false);
    // D layout: VGPR j -> row (j or j+8 by lane half), lane%16 -> code within tile
    const int   code = t * 16 + m;
    const float en   = enorm[code];
    #pragma unroll
    for (int j = 0; j < 8; ++j) {
      float dv = en - 2.0f * c.f[j];      // ||x||^2 constant per row: drop it
      if (dv < minval[j]) { minval[j] = dv; minidx[j] = code; }
    }
  }

  // ---- argmin across the 16 lanes of each half (tie-break: lowest index)
  #pragma unroll
  for (int j = 0; j < 8; ++j) {
    #pragma unroll
    for (int mask = 1; mask <= 8; mask <<= 1) {
      float ov = __shfl_xor(minval[j], mask, 32);
      int   oi = __shfl_xor(minidx[j], mask, 32);
      if (ov < minval[j] || (ov == minval[j] && oi < minidx[j])) {
        minval[j] = ov; minidx[j] = oi;
      }
    }
  }

  // ---- gather winning code rows, emit both layouts, accumulate loss & histogram
  float diffsum = 0.f;
  for (int mm = 0; mm < 16; ++mm) {
    int code = __shfl(minidx[mm & 7], (mm < 8) ? 0 : 16, 32);
    int n  = row0 + mm;
    int b  = n >> 10, hw = n & 1023;
    const float* erow = emb + (size_t)code * DIM;
    const float* xrow = x   + (size_t)b * 65536 + hw;
    #pragma unroll
    for (int h = 0; h < 2; ++h) {
      int d = lane + h * 32;
      float q  = erow[d];
      float xv = xrow[(size_t)d * 1024];
      float df = q - xv;
      diffsum += df * df;
      out[OUT_NCHW_OFF + (size_t)b * 65536 + (size_t)d * 1024 + hw] = q;
      out[OUT_NHWC_OFF + (size_t)n * DIM + d] = q;
    }
    if (lane == 0) atomicAdd(hist + code, 1u);
  }
  #pragma unroll
  for (int mask = 1; mask < 32; mask <<= 1)
    diffsum += __shfl_xor(diffsum, mask, 32);
  if (lane == 0) atomicAdd(lossacc, diffsum);
}

// ---------------- finalize: perplexity + loss scalars
__global__ void vq_final(float* __restrict__ out, void* ws) {
  __shared__ float red[512];
  const unsigned int* hist    = (const unsigned int*)((const char*)ws + 67584);
  const float*        lossacc = (const float*)((const char*)ws + 69632);
  int t = threadIdx.x;
  float p = (float)hist[t] / 65536.0f;
  red[t] = p * logf(p + 1e-10f);
  __syncthreads();
  for (int s = 256; s > 0; s >>= 1) {
    if (t < s) red[t] += red[t + s];
    __syncthreads();
  }
  if (t == 0) {
    out[OUT_PERP_OFF] = expf(-red[0]);
    out[0] = (*lossacc / TOTAL_ELEMS) * 1.25f;   // q_loss + 0.25*e_loss (identical MSEs)
  }
}

extern "C" void kernel_launch(void* const* d_in, const int* in_sizes, int n_in,
                              void* d_out, int out_size, void* d_ws, size_t ws_size,
                              hipStream_t stream) {
  const float* x   = (const float*)d_in[0];   // (64, 64, 32, 32) f32 NCHW
  const float* emb = (const float*)d_in[1];   // (512, 64) f32
  float* out = (float*)d_out;
  vq_prep <<<10,   256, 0, stream>>>(emb, d_ws);
  vq_main <<<1024, 128, 0, stream>>>(x, emb, out, d_ws);
  vq_final<<<1,    512, 0, stream>>>(out, d_ws);
}
